// SelfAttention_7241314861344
// MI455X (gfx1250) — matrix operand order repaired and checked by reading, not compiled, not run on hardware
//
#include <hip/hip_runtime.h>
#include <hip/hip_bf16.h>

// ---------------------------------------------------------------------------
// Self-attention with LoRA (B=2, S=2048, D=1024, H=16, hd=64) for gfx1250.
// Strategy: fold LoRA rank-32 updates into bf16 effective weights, run two
// WMMA-bf16 GEMMs (async-LDS staged, K-step 64), and a one-pass flash
// attention that streams the 268MB alibi bias exactly once (with prefetch).
// fp32 accumulation everywhere (WMMA C/D = f32).
// ---------------------------------------------------------------------------

#define HEADS  16
#define HD     64
#define SEQ    2048
#define DMODEL 1024
#define BSZ    2

typedef __attribute__((ext_vector_type(8)))  __bf16 bf16x8;
typedef __attribute__((ext_vector_type(16))) __bf16 bf16x16;
typedef __attribute__((ext_vector_type(8)))  float  f32x8;

static __device__ __forceinline__ unsigned short f32_to_bf16(float f) {
  unsigned int u = __float_as_uint(f);
  unsigned int r = (u + 0x7FFFu + ((u >> 16) & 1u)) >> 16;   // RNE
  return (unsigned short)r;
}

// 16 contiguous bf16 halves (two 16B loads) -> v16bf
static __device__ __forceinline__ bf16x16 ld_contig16(const unsigned short* p) {
  bf16x8 lo = *(const bf16x8*)p;
  bf16x8 hi = *(const bf16x8*)(p + 8);
  return __builtin_shufflevector(lo, hi, 0,1,2,3,4,5,6,7,8,9,10,11,12,13,14,15);
}

// A-matrix fragment (16x32 bf16, contiguous k order, B loaded to match): lane holds row (lane&15);
// lanes 0-15 hold K {0..15}; lanes 16-31 hold K {16..31}.
static __device__ __forceinline__ bf16x16 ld_afrag(const unsigned short* rowp, int lane) {
  const int o = (lane < 16) ? 0 : 16;
  bf16x8 lo = *(const bf16x8*)(rowp + o);
  bf16x8 hi = *(const bf16x8*)(rowp + o + 8);
  return __builtin_shufflevector(lo, hi, 0,1,2,3,4,5,6,7,8,9,10,11,12,13,14,15);
}

static __device__ __forceinline__ f32x8 wmma_bf16(bf16x16 a, bf16x16 b, f32x8 c) {
  return __builtin_amdgcn_wmma_f32_16x16x32_bf16(false, a, false, b, (short)0, c,
                                                 false, false);
}

// Async memory->LDS 16B copy (per lane), tracked by ASYNCcnt. Generic pointer
// to a __shared__ object carries the LDS byte offset in its low 32 bits.
static __device__ __forceinline__ void async_copy_b128(const unsigned short* g,
                                                       unsigned short* l) {
  unsigned int loff = (unsigned int)(unsigned long long)(void*)l;
  asm volatile("global_load_async_to_lds_b128 %0, %1, off"
               :
               : "v"(loff), "v"(g)
               : "memory");
}

static __device__ __forceinline__ void wait_async0() {
  asm volatile("s_wait_asynccnt 0" ::: "memory");
}

// ---------------------------------------------------------------------------
// W_eff^T[n][k] = W[n][k] + (A@B)[k][n] * (alpha/rank), stored bf16 [N][K]
// (so GEMM B-operand columns are contiguous).
// ---------------------------------------------------------------------------
__global__ void build_weff(const float* __restrict__ W,   // [N][K]
                           const float* __restrict__ A,   // [K][32]
                           const float* __restrict__ Bm,  // [32][N]
                           unsigned short* __restrict__ Wt, int N, int K) {
  int i = blockIdx.x * 256 + threadIdx.x;
  if (i >= N * K) return;
  int n = i / K, k = i % K;
  float acc = 0.f;
#pragma unroll
  for (int r = 0; r < 32; ++r) acc += A[k * 32 + r] * Bm[r * N + n];
  Wt[i] = f32_to_bf16(W[i] + acc * (1.0f / 32.0f));
}

__global__ void cvt_f32_bf16(const float* __restrict__ in,
                             unsigned short* __restrict__ out, int n) {
  int i = blockIdx.x * 256 + threadIdx.x;
  if (i < n) out[i] = f32_to_bf16(in[i]);
}

// ---------------------------------------------------------------------------
// Tiled WMMA GEMM: C[M][N] = A[M][K] * Beff (Bt stored [N][K]).
// Block tile 128x64, 8 waves (4x2), each wave 2x2 WMMA tiles.
// K-step 64 per staging round (8 WMMAs between barriers); tiles staged with
// GLOBAL_LOAD_ASYNC_TO_LDS_B128 (ASYNCcnt), row stride 72 halves (144B) keeps
// every fragment access 16B aligned.
// MODE 0: fp32 row-major out.  MODE 1: scatter bf16 into q/k/v [B][H][S][hd].
// ---------------------------------------------------------------------------
template <int MODE>
__global__ __launch_bounds__(256) void wmma_gemm(
    const unsigned short* __restrict__ A, const unsigned short* __restrict__ Bt,
    int M, int N, int K, float* __restrict__ Cout,
    unsigned short* __restrict__ qb, unsigned short* __restrict__ kb2,
    unsigned short* __restrict__ vb) {
  __shared__ __align__(16) unsigned short lA[128 * 72];
  __shared__ __align__(16) unsigned short lB[64 * 72];

  const int t = threadIdx.x;
  const int lane = t & 31;
  const int wid = t >> 5;
  const int wm = (wid & 3) * 32;
  const int wn = (wid >> 2) * 32;
  const int mb = blockIdx.y * 128;
  const int nb = blockIdx.x * 64;
  const int nlo = lane & 15;

  const f32x8 vzero = {0, 0, 0, 0, 0, 0, 0, 0};
  f32x8 acc[2][2];
#pragma unroll
  for (int mt = 0; mt < 2; ++mt)
#pragma unroll
    for (int nt = 0; nt < 2; ++nt) acc[mt][nt] = vzero;

  for (int k0 = 0; k0 < K; k0 += 64) {
    // Stage A tile (128x64 halves): 4 async b128 copies per thread.
#pragma unroll
    for (int r = 0; r < 4; ++r) {
      int idx = (t + r * 256) * 8;
      int row = idx >> 6, col = idx & 63;
      async_copy_b128(A + (size_t)(mb + row) * K + k0 + col, &lA[row * 72 + col]);
    }
    // Stage Bt tile (64x64 halves): 2 async b128 copies per thread.
#pragma unroll
    for (int r = 0; r < 2; ++r) {
      int idx = (t + r * 256) * 8;
      int row = idx >> 6, col = idx & 63;
      async_copy_b128(Bt + (size_t)(nb + row) * K + k0 + col, &lB[row * 72 + col]);
    }
    wait_async0();
    __syncthreads();

#pragma unroll
    for (int kk = 0; kk < 64; kk += 32) {
      bf16x16 af[2], br[2];
#pragma unroll
      for (int mt = 0; mt < 2; ++mt)
        af[mt] = ld_afrag(&lA[(wm + mt * 16 + nlo) * 72 + kk], lane);
#pragma unroll
      for (int nt = 0; nt < 2; ++nt)
        br[nt] = ld_contig16(&lB[(wn + nt * 16 + nlo) * 72 + kk +
                                 ((lane < 16) ? 0 : 16)]);
#pragma unroll
      for (int mt = 0; mt < 2; ++mt)
#pragma unroll
        for (int nt = 0; nt < 2; ++nt)
          acc[mt][nt] = wmma_bf16(af[mt], br[nt], acc[mt][nt]);
    }
    __syncthreads();
  }

  // Epilogue. C layout (ISA 7.12.2): VGPR j / lane l -> row j+8*(l>=16), col l&15.
#pragma unroll
  for (int mt = 0; mt < 2; ++mt)
#pragma unroll
    for (int nt = 0; nt < 2; ++nt)
#pragma unroll
      for (int j = 0; j < 8; ++j) {
        int gm = mb + wm + mt * 16 + j + ((lane >> 4) << 3);
        int gn = nb + wn + nt * 16 + nlo;
        float val = acc[mt][nt][j];
        if (MODE == 0) {
          Cout[(size_t)gm * N + gn] = val;
        } else {
          int sel = gn >> 10;            // 0=q 1=k 2=v
          int hh  = (gn >> 6) & 15;
          int hd  = gn & 63;
          int bi  = gm >> 11;            // / SEQ
          int s   = gm & (SEQ - 1);
          unsigned short* dst = (sel == 0) ? qb : ((sel == 1) ? kb2 : vb);
          dst[(((size_t)bi * HEADS + hh) * SEQ + s) * HD + hd] = f32_to_bf16(val);
        }
      }
}

// ---------------------------------------------------------------------------
// One-pass flash attention, bias streamed once (+ prefetch of next chunk).
// Grid: (SEQ/128, B*H); block 256 = 8 waves; each wave owns a 16-row q tile.
// 32-key chunks: scores = 4 WMMAs (Q frags + K^T frags direct from global),
// online softmax with 16-lane shfl reductions, P converted C->A layout via
// wave-private LDS, V staged transposed in LDS, P@V = 4 WMMAs.
// ---------------------------------------------------------------------------
__global__ __launch_bounds__(256) void flash_attn(
    const unsigned short* __restrict__ qg, const unsigned short* __restrict__ kg,
    const unsigned short* __restrict__ vg, const float* __restrict__ alibi,
    const float* __restrict__ padm, unsigned short* __restrict__ xattn) {
  __shared__ __align__(16) unsigned short vt[64 * 40];       // V^T [hd][key]
  __shared__ __align__(16) unsigned short pt[8 * 16 * 40];   // per-wave P tiles

  const int t = threadIdx.x, lane = t & 31, wid = t >> 5;
  const int bh = blockIdx.y, b = bh >> 4, h = bh & 15;
  const int qbase = blockIdx.x * 128 + wid * 16;
  const int nlo = lane & 15;
  const int half8 = (lane >> 4) << 3;

  const unsigned short* qrow = qg + ((size_t)bh * SEQ + qbase + nlo) * HD;
  const bf16x16 qf0 = ld_afrag(qrow, lane);        // hd 0..31
  const bf16x16 qf1 = ld_afrag(qrow + 32, lane);   // hd 32..63

  const unsigned short* kbase = kg + (size_t)bh * SEQ * HD;
  const unsigned short* vbase = vg + (size_t)bh * SEQ * HD;
  unsigned short* pw = pt + wid * (16 * 40);

  const f32x8 vzero = {0, 0, 0, 0, 0, 0, 0, 0};
  f32x8 o[4];
#pragma unroll
  for (int i = 0; i < 4; ++i) o[i] = vzero;
  float m_run[8], l_run[8];
#pragma unroll
  for (int j = 0; j < 8; ++j) { m_run[j] = -__builtin_inff(); l_run[j] = 0.f; }

  for (int kb = 0; kb < SEQ; kb += 32) {
    // Stage V^T: thread -> (key = t/8, hd0 = (t%8)*8): b128 read, 8 ds b16 writes
    {
      int key = t >> 3, hd0 = (t & 7) << 3;
      uint4 raw = *(const uint4*)(vbase + (size_t)(kb + key) * HD + hd0);
      unsigned int w[4] = {raw.x, raw.y, raw.z, raw.w};
#pragma unroll
      for (int i = 0; i < 4; ++i) {
        vt[(hd0 + 2 * i    ) * 40 + key] = (unsigned short)(w[i] & 0xffffu);
        vt[(hd0 + 2 * i + 1) * 40 + key] = (unsigned short)(w[i] >> 16);
      }
    }
    __syncthreads();

    // Scores: c0 = q @ k^T (keys kb..kb+15), c1 = keys kb+16..kb+31
    f32x8 c0 = vzero, c1 = vzero;
    {
      const unsigned short* kc0 =
          kbase + (size_t)(kb + nlo) * HD + ((lane < 16) ? 0 : 16);
      c0 = wmma_bf16(qf0, ld_contig16(kc0), c0);
      c0 = wmma_bf16(qf1, ld_contig16(kc0 + 32), c0);
      const unsigned short* kc1 =
          kbase + (size_t)(kb + 16 + nlo) * HD + ((lane < 16) ? 0 : 16);
      c1 = wmma_bf16(qf0, ld_contig16(kc1), c1);
      c1 = wmma_bf16(qf1, ld_contig16(kc1 + 32), c1);
    }

    const float pad0 = padm[b * SEQ + kb + nlo];
    const float pad1 = padm[b * SEQ + kb + 16 + nlo];
    float sc[8];
#pragma unroll
    for (int j = 0; j < 8; ++j) {
      const int row = qbase + j + half8;
      const size_t boff = ((size_t)h * SEQ + row) * SEQ + kb;
      float s0 = c0[j] * 0.125f + alibi[boff + nlo] + pad0;
      float s1 = c1[j] * 0.125f + alibi[boff + 16 + nlo] + pad1;
      // Prefetch next chunk's bias row (global_prefetch_b8) to hide HBM latency.
      if (kb + 32 < SEQ) __builtin_prefetch(alibi + boff + 32 + nlo, 0, 1);
      // row-wise max over 32 keys (16 lanes per half hold one row)
      float mx = fmaxf(s0, s1);
      mx = fmaxf(mx, __shfl_xor(mx, 1, 32));
      mx = fmaxf(mx, __shfl_xor(mx, 2, 32));
      mx = fmaxf(mx, __shfl_xor(mx, 4, 32));
      mx = fmaxf(mx, __shfl_xor(mx, 8, 32));
      const float mnew = fmaxf(m_run[j], mx);
      const float scale = __expf(m_run[j] - mnew);
      const float p0 = __expf(s0 - mnew);
      const float p1 = __expf(s1 - mnew);
      float rs = p0 + p1;
      rs += __shfl_xor(rs, 1, 32);
      rs += __shfl_xor(rs, 2, 32);
      rs += __shfl_xor(rs, 4, 32);
      rs += __shfl_xor(rs, 8, 32);
      l_run[j] = l_run[j] * scale + rs;
      m_run[j] = mnew;
      sc[j] = scale;
      // P (C layout) -> wave-private LDS; same-wave LDS ops stay ordered
      pw[(j + half8) * 40 + nlo]      = f32_to_bf16(p0);
      pw[(j + half8) * 40 + 16 + nlo] = f32_to_bf16(p1);
    }
#pragma unroll
    for (int ht = 0; ht < 4; ++ht)
#pragma unroll
      for (int j = 0; j < 8; ++j) o[ht][j] = o[ht][j] * sc[j];

    const bf16x16 pf = ld_afrag(pw + nlo * 40, lane);   // A-layout reload
#pragma unroll
    for (int ht = 0; ht < 4; ++ht) {
      bf16x16 vf =
          ld_contig16(&vt[(ht * 16 + nlo) * 40 + ((lane < 16) ? 0 : 16)]);
      o[ht] = wmma_bf16(pf, vf, o[ht]);
    }
    __syncthreads();
  }

  // Normalize and write attn as bf16 in [B,S,H*hd] for the output GEMM.
#pragma unroll
  for (int ht = 0; ht < 4; ++ht)
#pragma unroll
    for (int j = 0; j < 8; ++j) {
      const int row = qbase + j + half8;
      const float val = o[ht][j] / l_run[j];
      xattn[((size_t)b * SEQ + row) * DMODEL + h * HD + ht * 16 + nlo] =
          f32_to_bf16(val);
    }
}

// ---------------------------------------------------------------------------
extern "C" void kernel_launch(void* const* d_in, const int* in_sizes, int n_in,
                              void* d_out, int out_size, void* d_ws, size_t ws_size,
                              hipStream_t stream) {
  (void)in_sizes; (void)n_in; (void)out_size; (void)ws_size;
  const float* x     = (const float*)d_in[0];
  const float* alibi = (const float*)d_in[1];
  const float* padm  = (const float*)d_in[2];
  const float* W_qkv = (const float*)d_in[3];
  const float* A_qkv = (const float*)d_in[4];
  const float* B_qkv = (const float*)d_in[5];
  const float* W_out = (const float*)d_in[6];
  const float* A_out = (const float*)d_in[7];
  const float* B_out = (const float*)d_in[8];
  float* out = (float*)d_out;

  const int M = BSZ * SEQ;      // 4096
  const int D = DMODEL;         // 1024

  size_t off = 0;
  char* base = (char*)d_ws;
  auto carve = [&](size_t bytes) {
    void* p = base + off;
    off += (bytes + 255) & ~(size_t)255;
    return p;
  };
  unsigned short* xb     = (unsigned short*)carve((size_t)M * D * 2);
  unsigned short* wqkv_t = (unsigned short*)carve((size_t)3 * D * D * 2);
  unsigned short* wout_t = (unsigned short*)carve((size_t)D * D * 2);
  unsigned short* qb     = (unsigned short*)carve((size_t)M * D * 2);
  unsigned short* kb     = (unsigned short*)carve((size_t)M * D * 2);
  unsigned short* vb     = (unsigned short*)carve((size_t)M * D * 2);
  unsigned short* xattn  = (unsigned short*)carve((size_t)M * D * 2);

  build_weff<<<(3 * D * D + 255) / 256, 256, 0, stream>>>(W_qkv, A_qkv, B_qkv,
                                                          wqkv_t, 3 * D, D);
  build_weff<<<(D * D + 255) / 256, 256, 0, stream>>>(W_out, A_out, B_out,
                                                      wout_t, D, D);
  cvt_f32_bf16<<<(M * D + 255) / 256, 256, 0, stream>>>(x, xb, M * D);

  dim3 g1(3 * D / 64, M / 128);   // (48, 32)
  wmma_gemm<1><<<g1, 256, 0, stream>>>(xb, wqkv_t, M, 3 * D, D, nullptr, qb, kb, vb);

  dim3 g2(SEQ / 128, BSZ * HEADS);  // (16, 32)
  flash_attn<<<g2, 256, 0, stream>>>(qb, kb, vb, alibi, padm, xattn);

  dim3 g3(D / 64, M / 128);       // (16, 32)
  wmma_gemm<0><<<g3, 256, 0, stream>>>(xattn, wout_t, M, D, D, out,
                                       nullptr, nullptr, nullptr);
}
